// MinkowskiFCNN_46832323395840
// MI455X (gfx1250) — compile-verified, matrix operand order checked by reading, not run on hardware
//
#include <hip/hip_runtime.h>
#include <cstddef>
#include <cstdint>

#define G3D 512

typedef __attribute__((ext_vector_type(16))) __bf16 v16bf;
typedef __attribute__((ext_vector_type(8)))  float  v8f;

__device__ __forceinline__ unsigned short f2bf(float f) {
  unsigned u = __builtin_bit_cast(unsigned, f);
  unsigned r = u + 0x7FFFu + ((u >> 16) & 1u);   // round-to-nearest-even
  return (unsigned short)(r >> 16);
}

// ---------------- elementwise / BN kernels ----------------

__global__ void k_outer(const float* __restrict__ f, const float* __restrict__ w,
                        long n, int C, float* __restrict__ out) {
  long t = (long)blockIdx.x * blockDim.x + threadIdx.x;
  if (t >= n) return;
  out[t] = f[t / C] * w[t % C];
}

__global__ void k_colstats(const float* __restrict__ x, int M, int C,
                           float* __restrict__ mu, float* __restrict__ var) {
  __shared__ float s1[256], s2[256];
  int c = blockIdx.x;
  float a = 0.f, b = 0.f;
  for (int m = threadIdx.x; m < M; m += 256) {
    float v = x[(size_t)m * C + c];
    a += v; b += v * v;
  }
  s1[threadIdx.x] = a; s2[threadIdx.x] = b;
  __syncthreads();
  for (int s = 128; s > 0; s >>= 1) {
    if ((int)threadIdx.x < s) { s1[threadIdx.x] += s1[threadIdx.x + s];
                                s2[threadIdx.x] += s2[threadIdx.x + s]; }
    __syncthreads();
  }
  if (threadIdx.x == 0) {
    float m1 = s1[0] / (float)M;
    mu[c] = m1;
    var[c] = s2[0] / (float)M - m1 * m1;
  }
}

__global__ void k_bnact(const float* __restrict__ x, long n, int C,
                        const float* __restrict__ mu, const float* __restrict__ var,
                        const float* __restrict__ g, const float* __restrict__ b,
                        float* __restrict__ y) {
  long t = (long)blockIdx.x * blockDim.x + threadIdx.x;
  if (t >= n) return;
  int c = (int)(t % C);
  float v = (x[t] - mu[c]) * rsqrtf(var[c] + 1e-5f) * g[c] + b[c];
  y[t] = v >= 0.f ? v : 0.01f * v;
}

__global__ void k_cvt(const float* __restrict__ x, int C, int Kp, long n,
                      unsigned short* __restrict__ y) {
  long t = (long)blockIdx.x * blockDim.x + threadIdx.x;
  if (t >= n) return;
  int c = (int)(t % Kp);
  long m = t / Kp;
  y[t] = (c < C) ? f2bf(x[m * (long)C + c]) : (unsigned short)0;
}

// Swizzle weights (27, Cin, Cout) f32 -> WMMA B-fragment layout bf16:
// Wt[off][kb][nt][col 0..15][kr 0..31], kr contiguous per column so each
// lane's fragment is two 16B contiguous loads.
__global__ void k_wpadt(const float* __restrict__ W, int Cin, int Cout,
                        int ktiles, int ntiles, long n,
                        unsigned short* __restrict__ Wt) {
  long t = (long)blockIdx.x * blockDim.x + threadIdx.x;
  if (t >= n) return;
  int kr  = (int)(t % 32);
  int col = (int)((t / 32) % 16);
  int nt  = (int)((t / 512) % ntiles);
  int kb  = (int)((t / (512L * ntiles)) % ktiles);
  int off = (int)(t / (512L * ntiles * ktiles));
  int K = kb * 32 + kr;
  int c = nt * 16 + col;
  Wt[t] = (K < Cin) ? f2bf(W[((long)off * Cin + K) * Cout + c]) : (unsigned short)0;
}

__global__ void k_fill16(unsigned short* __restrict__ p, long n) {
  long t = (long)blockIdx.x * blockDim.x + threadIdx.x;
  if (t < n) p[t] = 0;
}

// ---------------- key build / bitonic sort / query ----------------

__global__ void k_keys(const int* __restrict__ c, int M, int stride, int Gs,
                       unsigned* __restrict__ ks, int* __restrict__ pm, int P2) {
  int i = blockIdx.x * blockDim.x + threadIdx.x;
  if (i >= P2) return;
  if (i < M) {
    unsigned b = (unsigned)c[4 * i];
    unsigned x = (unsigned)(c[4 * i + 1] / stride);
    unsigned y = (unsigned)(c[4 * i + 2] / stride);
    unsigned z = (unsigned)(c[4 * i + 3] / stride);
    ks[i] = ((b * Gs + x) * Gs + y) * Gs + z;
    pm[i] = i;
  } else {
    ks[i] = 0xFFFFFFFFu;
    pm[i] = 0;
  }
}

__global__ void k_bitonic(unsigned* __restrict__ ks, int* __restrict__ pm,
                          int j, int k, int n) {
  int i = blockIdx.x * blockDim.x + threadIdx.x;
  if (i >= n) return;
  int ixj = i ^ j;
  if (ixj <= i) return;
  bool up = ((i & k) == 0);
  unsigned a = ks[i], b = ks[ixj];
  if (up ? (a > b) : (a < b)) {
    ks[i] = b; ks[ixj] = a;
    int pa = pm[i]; pm[i] = pm[ixj]; pm[ixj] = pa;
  }
}

__global__ void k_query27(const int* __restrict__ oc, int Mout,
                          const unsigned* __restrict__ ks, const int* __restrict__ pm,
                          int Min, int stride, int Gs, int* __restrict__ idx) {
  long t = (long)blockIdx.x * blockDim.x + threadIdx.x;
  if (t >= (long)Mout * 27) return;
  int m = (int)(t / 27), o = (int)(t % 27);
  int dx = o / 9 - 1, dy = (o / 3) % 3 - 1, dz = o % 3 - 1;
  int b = oc[4 * m];
  int q0 = oc[4 * m + 1] + dx * stride;
  int q1 = oc[4 * m + 2] + dy * stride;
  int q2 = oc[4 * m + 3] + dz * stride;
  bool valid = (q0 >= 0 && q0 < G3D && q1 >= 0 && q1 < G3D && q2 >= 0 && q2 < G3D);
  int c0 = (q0 < 0 ? 0 : (q0 > G3D - 1 ? G3D - 1 : q0)) / stride;
  int c1 = (q1 < 0 ? 0 : (q1 > G3D - 1 ? G3D - 1 : q1)) / stride;
  int c2 = (q2 < 0 ? 0 : (q2 > G3D - 1 ? G3D - 1 : q2)) / stride;
  unsigned qk = (((unsigned)b * Gs + c0) * Gs + c1) * Gs + c2;
  int lo = 0, hi = Min;
  while (lo < hi) { int mid = (lo + hi) >> 1; if (ks[mid] < qk) lo = mid + 1; else hi = mid; }
  int pos = lo < Min ? lo : Min - 1;
  idx[t] = (valid && ks[pos] == qk) ? pm[pos] : -1;
}

__global__ void k_slice(const int* __restrict__ fc, int Nf,
                        const unsigned* __restrict__ ks, const int* __restrict__ pm,
                        int Min, int stride, int Gs,
                        const float* __restrict__ src, int C,
                        unsigned short* __restrict__ dst, int Kp, int colOff) {
  long t = (long)blockIdx.x * blockDim.x + threadIdx.x;
  if (t >= (long)Nf * C) return;
  int m = (int)(t / C), c = (int)(t % C);
  unsigned b = (unsigned)fc[4 * m];
  unsigned x = (unsigned)(fc[4 * m + 1] / stride);
  unsigned y = (unsigned)(fc[4 * m + 2] / stride);
  unsigned z = (unsigned)(fc[4 * m + 3] / stride);
  unsigned qk = ((b * Gs + x) * Gs + y) * Gs + z;
  int lo = 0, hi = Min;
  while (lo < hi) { int mid = (lo + hi) >> 1; if (ks[mid] < qk) lo = mid + 1; else hi = mid; }
  int pos = lo < Min ? lo : Min - 1;
  int gi = pm[pos];
  dst[(long)m * Kp + colOff + c] = f2bf(src[(long)gi * C + c]);
}

__global__ void k_maxpool(const float* __restrict__ in, const int* __restrict__ idx,
                          int Mout, int C, float* __restrict__ out) {
  long t = (long)blockIdx.x * blockDim.x + threadIdx.x;
  if (t >= (long)Mout * C) return;
  int m = (int)(t / C), c = (int)(t % C);
  float v = -3e38f;
  for (int o = 0; o < 27; ++o) {
    int gi = idx[(long)m * 27 + o];
    if (gi >= 0) { float f = in[(long)gi * C + c]; v = f > v ? f : v; }
  }
  out[t] = v;
}

// ---------------- gather + WMMA conv kernel ----------------
// 128 threads = 4 wave32s. Wave w owns rows [bx*64+16w, +16) and NT 16-col
// output tiles. A fragment: per-lane gather, row = lane&15, half = lane>>4
// selects the two contiguous 8-elem K runs (ISA 16-bit A layout) -> two
// global_load_b128. B fragments come from the pre-swizzled weight buffer as
// two contiguous global_load_b128 per tile (L2-resident, reused across M).
// Wave-uniform __any() skips taps where all 16 rows miss.
template <int NT>
__global__ __launch_bounds__(128) void k_conv_wmma(
    const unsigned short* __restrict__ Abf, int Kp,
    const unsigned short* __restrict__ Wt,    // (27, ktiles, ntiles, 16, 32)
    const int* __restrict__ idx,              // (Mout, 27)
    int Mout, int Cout, float* __restrict__ out) {
  const int lane = threadIdx.x & 31;
  const int wave = threadIdx.x >> 5;
  const int half = lane >> 4;
  const int l16  = lane & 15;
  const int mbase = blockIdx.x * 64 + wave * 16;
  const int nt0   = blockIdx.y * NT;          // first 16-col tile
  const int m = mbase + l16;
  const int ktiles = Kp >> 5;
  const int ntiles = Cout >> 4;
  v8f acc[NT];
#pragma unroll
  for (int i = 0; i < NT; ++i) acc[i] = (v8f){};

  int gi = (m < Mout) ? idx[(long)m * 27] : -1;
  for (int off = 0; off < 27; ++off) {
    // prefetch next tap's gathered row while computing this one
    int gnext = (off + 1 < 27 && m < Mout) ? idx[(long)m * 27 + off + 1] : -1;
    if (gnext >= 0)
      __builtin_prefetch(Abf + (size_t)gnext * Kp, 0, 1);

    if (__any(gi >= 0)) {                      // wave-uniform: skip all-miss taps
      for (int kb = 0; kb < ktiles; ++kb) {
        union { v16bf v; uint4 q[2]; } af;
        if (gi >= 0) {
          const unsigned short* ap = Abf + (size_t)gi * Kp + kb * 32 + half * 8;
          af.q[0] = *(const uint4*)ap;         // K = kb*32 + half*8 + 0..7
          af.q[1] = *(const uint4*)(ap + 16);  // K = kb*32 + 16 + half*8 + 0..7
        } else {
          af.q[0] = uint4{0u, 0u, 0u, 0u};
          af.q[1] = uint4{0u, 0u, 0u, 0u};
        }
        const unsigned short* wbase =
            Wt + (((size_t)off * ktiles + kb) * ntiles + nt0) * 512;
#pragma unroll
        for (int i = 0; i < NT; ++i) {
          union { v16bf v; uint4 q[2]; } bf;
          const unsigned short* wp = wbase + (size_t)i * 512 + l16 * 32;
          bf.q[0] = *(const uint4*)(wp + half * 8);
          bf.q[1] = *(const uint4*)(wp + 16 + half * 8);
          acc[i] = __builtin_amdgcn_wmma_f32_16x16x32_bf16(
              false, af.v, false, bf.v, (short)0, acc[i], false, false);
        }
      }
    }
    gi = gnext;
  }

#pragma unroll
  for (int i = 0; i < NT; ++i) {
    int col = (nt0 + i) * 16 + l16;
#pragma unroll
    for (int j = 0; j < 8; ++j) {
      int row = mbase + half * 8 + j;
      if (row < Mout) out[(size_t)row * Cout + col] = acc[i][j];
    }
  }
}

// ---------------- host orchestration ----------------

namespace {
struct Bump {
  char* p; size_t off, cap;
  void* take(size_t bytes) {
    size_t a = (off + 255) & ~(size_t)255;
    off = a + bytes;
    return p + a;
  }
};
}

extern "C" void kernel_launch(void* const* d_in, const int* in_sizes, int n_in,
                              void* d_out, int out_size, void* d_ws, size_t ws_size,
                              hipStream_t stream) {
  (void)n_in; (void)out_size;
  const float* feats = (const float*)d_in[0];
  const int*   coords= (const int*)d_in[1];
  const float* w_mlp = (const float*)d_in[2];
  const float* gm  = (const float*)d_in[3];  const float* bm  = (const float*)d_in[4];
  const float* W1  = (const float*)d_in[5];  const float* g1  = (const float*)d_in[6];  const float* b1  = (const float*)d_in[7];
  const float* W2  = (const float*)d_in[8];  const float* g2  = (const float*)d_in[9];  const float* b2  = (const float*)d_in[10];
  const float* W3  = (const float*)d_in[11]; const float* g3  = (const float*)d_in[12]; const float* b3  = (const float*)d_in[13];
  const float* W4  = (const float*)d_in[14]; const float* g4  = (const float*)d_in[15]; const float* b4  = (const float*)d_in[16];
  const float* W5a = (const float*)d_in[17]; const float* g5a = (const float*)d_in[18]; const float* b5a = (const float*)d_in[19];
  const float* W5b = (const float*)d_in[20]; const float* g5b = (const float*)d_in[21]; const float* b5b = (const float*)d_in[22];
  const float* W5c = (const float*)d_in[23]; const float* g5c = (const float*)d_in[24]; const float* b5c = (const float*)d_in[25];
  const int* c2   = (const int*)d_in[26]; const int* c4  = (const int*)d_in[27];
  const int* c8   = (const int*)d_in[28]; const int* c16 = (const int*)d_in[29];
  const int* c32  = (const int*)d_in[30]; const int* c64 = (const int*)d_in[31];
  const int* c128 = (const int*)d_in[32];

  const int N    = in_sizes[0];
  const int n2   = in_sizes[26] / 4, n4  = in_sizes[27] / 4, n8   = in_sizes[28] / 4;
  const int n16  = in_sizes[29] / 4, n32 = in_sizes[30] / 4, n64  = in_sizes[31] / 4;
  const int n128 = in_sizes[32] / 4;

  Bump ws{(char*)d_ws, 0, ws_size};

  auto L = [&](long n) { return dim3((unsigned)((n + 255) / 256)); };

  auto sortlvl = [&](const int* c, int M, int stride, unsigned** ks, int** pm) {
    int P2 = 1; while (P2 < M) P2 <<= 1;
    *ks = (unsigned*)ws.take((size_t)P2 * 4);
    *pm = (int*)ws.take((size_t)P2 * 4);
    int Gs = G3D / stride;
    k_keys<<<L(P2), 256, 0, stream>>>(c, M, stride, Gs, *ks, *pm, P2);
    for (int k = 2; k <= P2; k <<= 1)
      for (int j = k >> 1; j > 0; j >>= 1)
        k_bitonic<<<L(P2), 256, 0, stream>>>(*ks, *pm, j, k, P2);
  };

  auto query = [&](const int* oc, int Mout, unsigned* ks, int* pm, int Min, int stride) {
    int* idx = (int*)ws.take((size_t)Mout * 27 * 4);
    k_query27<<<L((long)Mout * 27), 256, 0, stream>>>(oc, Mout, ks, pm, Min,
                                                      stride, G3D / stride, idx);
    return idx;
  };

  auto conv = [&](const unsigned short* xb, int Kp, const int* ic, int Min,
                  const int* oc, int Mout, const float* W, int Cin, int Cout,
                  int stride, float* out) {
    size_t mark = ws.off;
    unsigned* ks; int* pm; sortlvl(ic, Min, stride, &ks, &pm);
    int* idx = query(oc, Mout, ks, pm, Min, stride);
    const int ktiles = Kp / 32, ntiles = Cout / 16;
    long wn = 27L * ktiles * ntiles * 512;
    unsigned short* Wt = (unsigned short*)ws.take((size_t)wn * 2);
    k_wpadt<<<L(wn), 256, 0, stream>>>(W, Cin, Cout, ktiles, ntiles, wn, Wt);
    const int NT = (ntiles % 4 == 0) ? 4 : 3;
    dim3 g((unsigned)((Mout + 63) / 64), (unsigned)(ntiles / NT));
    if (NT == 4)
      k_conv_wmma<4><<<g, 128, 0, stream>>>(xb, Kp, Wt, idx, Mout, Cout, out);
    else
      k_conv_wmma<3><<<g, 128, 0, stream>>>(xb, Kp, Wt, idx, Mout, Cout, out);
    ws.off = mark;
  };

  auto bn = [&](const float* x, int M, int C, const float* g, const float* b, float* dst) {
    size_t mark = ws.off;
    float* mu  = (float*)ws.take((size_t)C * 4);
    float* var = (float*)ws.take((size_t)C * 4);
    k_colstats<<<dim3((unsigned)C), 256, 0, stream>>>(x, M, C, mu, var);
    k_bnact<<<L((long)M * C), 256, 0, stream>>>(x, (long)M * C, C, mu, var, g, b, dst);
    ws.off = mark;
  };

  auto pool = [&](const float* in, const int* ic, int Min, const int* oc, int Mout,
                  int C, int stride, float* out) {
    size_t mark = ws.off;
    unsigned* ks; int* pm; sortlvl(ic, Min, stride, &ks, &pm);
    int* idx = query(oc, Mout, ks, pm, Min, stride);
    k_maxpool<<<L((long)Mout * C), 256, 0, stream>>>(in, idx, Mout, C, out);
    ws.off = mark;
  };

  auto slice = [&](const float* lf, const int* lc, int Ml, int C, int stride,
                   unsigned short* dst, int Kp, int colOff) {
    size_t mark = ws.off;
    unsigned* ks; int* pm; sortlvl(lc, Ml, stride, &ks, &pm);
    k_slice<<<L((long)N * C), 256, 0, stream>>>(coords, N, ks, pm, Ml, stride,
                                                G3D / stride, lf, C, dst, Kp, colOff);
    ws.off = mark;
  };

  auto cvt = [&](const float* x, int M, int C, int Kp, unsigned short* dst) {
    k_cvt<<<L((long)M * Kp), 256, 0, stream>>>(x, C, Kp, (long)M * Kp, dst);
  };

  // ---- persistent buffers ----
  unsigned short* xb   = (unsigned short*)ws.take((size_t)N * 32 * 2);
  float*          y1   = (float*)ws.take((size_t)n2 * 48 * 4);
  unsigned short* y1b  = (unsigned short*)ws.take((size_t)n2 * 64 * 2);
  float*          y2   = (float*)ws.take((size_t)n8 * 64 * 4);
  unsigned short* y2b  = (unsigned short*)ws.take((size_t)n8 * 64 * 2);
  float*          y3   = (float*)ws.take((size_t)n32 * 96 * 4);
  unsigned short* y3b  = (unsigned short*)ws.take((size_t)n32 * 96 * 2);
  float*          y4   = (float*)ws.take((size_t)n128 * 128 * 4);
  unsigned short* xcb  = (unsigned short*)ws.take((size_t)N * 352 * 2);   // Kp(336)=352
  unsigned short* y5ab = (unsigned short*)ws.take((size_t)n2 * 256 * 2);
  unsigned short* y5bb = (unsigned short*)ws.take((size_t)n4 * 512 * 2);

  // ---- stage 0: MLP 1->32 + BN ----
  {
    size_t mark = ws.off;
    float* t0 = (float*)ws.take((size_t)N * 32 * 4);
    k_outer<<<L((long)N * 32), 256, 0, stream>>>(feats, w_mlp, (long)N * 32, 32, t0);
    bn(t0, N, 32, gm, bm, t0);
    cvt(t0, N, 32, 32, xb);
    ws.off = mark;
  }
  // ---- conv1 32->48 @ coords, pool -> c2 ----
  {
    size_t mark = ws.off;
    float* a = (float*)ws.take((size_t)N * 48 * 4);
    conv(xb, 32, coords, N, coords, N, W1, 32, 48, 1, a);
    bn(a, N, 48, g1, b1, a);
    pool(a, coords, N, c2, n2, 48, 1, y1);
    ws.off = mark;
  }
  cvt(y1, n2, 48, 64, y1b);
  // ---- conv2 48->64 c2->c4, pool c4->c8 ----
  {
    size_t mark = ws.off;
    float* a = (float*)ws.take((size_t)n4 * 64 * 4);
    conv(y1b, 64, c2, n2, c4, n4, W2, 48, 64, 2, a);
    bn(a, n4, 64, g2, b2, a);
    pool(a, c4, n4, c8, n8, 64, 4, y2);
    ws.off = mark;
  }
  cvt(y2, n8, 64, 64, y2b);
  // ---- conv3 64->96 c8->c16, pool c16->c32 ----
  {
    size_t mark = ws.off;
    float* a = (float*)ws.take((size_t)n16 * 96 * 4);
    conv(y2b, 64, c8, n8, c16, n16, W3, 64, 96, 8, a);
    bn(a, n16, 96, g3, b3, a);
    pool(a, c16, n16, c32, n32, 96, 16, y3);
    ws.off = mark;
  }
  cvt(y3, n32, 96, 96, y3b);
  // ---- conv4 96->128 c32->c64, pool c64->c128 ----
  {
    size_t mark = ws.off;
    float* a = (float*)ws.take((size_t)n64 * 128 * 4);
    conv(y3b, 96, c32, n32, c64, n64, W4, 96, 128, 32, a);
    bn(a, n64, 128, g4, b4, a);
    pool(a, c64, n64, c128, n128, 128, 64, y4);
    ws.off = mark;
  }
  // ---- concat via slice-gathers into bf16 (N, 352) ----
  k_fill16<<<L((long)N * 352), 256, 0, stream>>>(xcb, (long)N * 352);
  slice(y1, c2,   n2,   48,   2, xcb, 352, 0);
  slice(y2, c8,   n8,   64,   8, xcb, 352, 48);
  slice(y3, c32,  n32,  96,  32, xcb, 352, 112);
  slice(y4, c128, n128, 128, 128, xcb, 352, 208);
  // ---- conv5a 336->256 coords->c2 ----
  {
    size_t mark = ws.off;
    float* a = (float*)ws.take((size_t)n2 * 256 * 4);
    conv(xcb, 352, coords, N, c2, n2, W5a, 336, 256, 1, a);
    bn(a, n2, 256, g5a, b5a, a);
    cvt(a, n2, 256, 256, y5ab);
    ws.off = mark;
  }
  // ---- conv5b 256->512 c2->c4 ----
  {
    size_t mark = ws.off;
    float* a = (float*)ws.take((size_t)n4 * 512 * 4);
    conv(y5ab, 256, c2, n2, c4, n4, W5b, 256, 512, 2, a);
    bn(a, n4, 512, g5b, b5b, a);
    cvt(a, n4, 512, 512, y5bb);
    ws.off = mark;
  }
  // ---- conv5c 512->1024 c4->c8, BN -> d_out ----
  {
    size_t mark = ws.off;
    float* a = (float*)ws.take((size_t)n8 * 1024 * 4);
    conv(y5bb, 512, c4, n4, c8, n8, W5c, 512, 1024, 4, a);
    bn(a, n8, 1024, g5c, b5c, (float*)d_out);
    ws.off = mark;
  }
}
